// CalderaQuantizedLinear_75642964017925
// MI455X (gfx1250) — compile-verified, matrix operand order checked by reading, not run on hardware
//
#include <hip/hip_runtime.h>
#include <hip/hip_bf16.h>
#include <stdint.h>

typedef __bf16 bf16;
typedef __attribute__((ext_vector_type(16))) __bf16 v16bf;
typedef __attribute__((ext_vector_type(8)))  float  v8f;

#define ROWS   8192   // 4*2048
#define NCOLS  4096   // N == M
#define RANKT  128    // fused rank (64 quantized + 64 fine-tune)
#define TM     128
#define TN     128
#define KC     64
#define LDST   72     // 64 + 8 bf16 padding to spread LDS banks

union FragU { v16bf bf; uint4 u[2]; };
union AccU  { v8f v; float f[8]; };
union Pack8 { uint4 u; bf16 b[8]; };

// ---------------------------------------------------------------------------
// Kernel 1: xh = FWHT(x * SU) / 64, stored bf16.  One 4096-elem row per block.
// ---------------------------------------------------------------------------
__global__ __launch_bounds__(256)
void caldera_fwht_pre(const float* __restrict__ x, const float* __restrict__ SU,
                      bf16* __restrict__ xh) {
  __shared__ float s[NCOLS];
  const int row = blockIdx.x;
  const float* xr = x + (size_t)row * NCOLS;
  for (int i = threadIdx.x; i < NCOLS; i += 256) s[i] = xr[i] * SU[i];
  __syncthreads();
  for (int h = 1; h < NCOLS; h <<= 1) {
    for (int p = threadIdx.x; p < (NCOLS >> 1); p += 256) {
      int i = ((p & ~(h - 1)) << 1) | (p & (h - 1));
      float a = s[i], b = s[i + h];
      s[i] = a + b;
      s[i + h] = a - b;
    }
    __syncthreads();
  }
  const float sc = 0.015625f;  // 1/sqrt(4096)
  bf16* o = xh + (size_t)row * NCOLS;
  for (int i = threadIdx.x; i < NCOLS; i += 256) o[i] = (bf16)(s[i] * sc);
}

// ---------------------------------------------------------------------------
// Kernel 5: out = FWHT(out) / 64 * SV * GLOBAL_SCALE, in place on f32 rows.
// ---------------------------------------------------------------------------
__global__ __launch_bounds__(256)
void caldera_fwht_post(float* __restrict__ out, const float* __restrict__ SV) {
  __shared__ float s[NCOLS];
  const int row = blockIdx.x;
  float* orow = out + (size_t)row * NCOLS;
  for (int i = threadIdx.x; i < NCOLS; i += 256) s[i] = orow[i];
  __syncthreads();
  for (int h = 1; h < NCOLS; h <<= 1) {
    for (int p = threadIdx.x; p < (NCOLS >> 1); p += 256) {
      int i = ((p & ~(h - 1)) << 1) | (p & (h - 1));
      float a = s[i], b = s[i + h];
      s[i] = a + b;
      s[i + h] = a - b;
    }
    __syncthreads();
  }
  const float sc = 0.015625f;  // 1/sqrt(4096), GLOBAL_SCALE == 1.0
  for (int i = threadIdx.x; i < NCOLS; i += 256) orow[i] = s[i] * sc * SV[i];
}

// ---------------------------------------------------------------------------
// Dequant: each thread expands one codebook index into 8 bf16 values.
// ---------------------------------------------------------------------------
__global__ __launch_bounds__(256)
void caldera_dequant8(const int* __restrict__ idxs, const float* __restrict__ grid,
                      bf16* __restrict__ out, float scale, int total) {
  int t = blockIdx.x * 256 + threadIdx.x;
  if (t >= total) return;
  int idx = idxs[t];
  const float4* g = (const float4*)(grid + (size_t)idx * 8);
  float4 lo = g[0], hi = g[1];
  Pack8 p;
  p.b[0] = (bf16)(lo.x * scale); p.b[1] = (bf16)(lo.y * scale);
  p.b[2] = (bf16)(lo.z * scale); p.b[3] = (bf16)(lo.w * scale);
  p.b[4] = (bf16)(hi.x * scale); p.b[5] = (bf16)(hi.y * scale);
  p.b[6] = (bf16)(hi.z * scale); p.b[7] = (bf16)(hi.w * scale);
  *(uint4*)(out + (size_t)t * 8) = p.u;
}

__global__ __launch_bounds__(256)
void caldera_f32_to_bf16(const float* __restrict__ in, bf16* __restrict__ out, int n) {
  int t = blockIdx.x * 256 + threadIdx.x;
  if (t < n) out[t] = (bf16)in[t];
}

// WclT[n][r] = (r < 64) ? W_l[r][n]*L_SCALE : L_ft[n][r-64]
__global__ __launch_bounds__(256)
void caldera_build_wclt(const int* __restrict__ L_idxs, const float* __restrict__ grid,
                        const float* __restrict__ L_ft, bf16* __restrict__ WclT) {
  int t = blockIdx.x * 256 + threadIdx.x;  // over 4096*128
  int n = t >> 7;
  int r = t & 127;
  float v;
  if (r < 64) {
    int idx = L_idxs[r * (NCOLS / 8) + (n >> 3)];
    v = grid[(size_t)idx * 8 + (n & 7)] * 1.1f;  // L_SCALE
  } else {
    v = L_ft[(size_t)n * 64 + (r - 64)];
  }
  WclT[(size_t)n * RANKT + r] = (bf16)v;
}

// ---------------------------------------------------------------------------
// Async staging: 128x64 bf16 tile, memory -> LDS via GLOBAL_LOAD_ASYNC_TO_LDS.
// Each of 256 threads issues 4x b128 async loads (1024 16B chunks total).
// Tracked by ASYNCcnt; no VGPR data round-trip, no ds_store on the data path.
// ---------------------------------------------------------------------------
__device__ __forceinline__ void issue_async_tile(const bf16* __restrict__ g, int ld,
                                                 int row0, int k0,
                                                 const bf16* lds_dst) {
  const int tid = threadIdx.x;
#pragma unroll
  for (int i = 0; i < 4; ++i) {
    int c = tid + i * 256;
    int r = c >> 3;
    int cc = c & 7;
    uint64_t ga = (uint64_t)(uintptr_t)(g + (size_t)(row0 + r) * ld + k0 + cc * 8);
    // LDS aperture maps addr[31:0] -> LDS byte offset
    uint32_t la = (uint32_t)(uintptr_t)(lds_dst + r * LDST + cc * 8);
    asm volatile("global_load_async_to_lds_b128 %0, %1, off"
                 :: "v"(la), "v"(ga) : "memory");
  }
}

__device__ __forceinline__ void wait_asynccnt0() {
  asm volatile("s_wait_asynccnt 0x0" ::: "memory");
}

__device__ __forceinline__ void mma_chunk(const bf16* As, const bf16* Bs,
                                          v8f acc[4][2], int wm, int wn,
                                          int half, int l16) {
#pragma unroll
  for (int ks = 0; ks < 2; ++ks) {
    const int kb = ks * 32 + half * 8;
    v16bf a[4], b[2];
#pragma unroll
    for (int mt = 0; mt < 4; ++mt) {
      const bf16* p = As + (wm * 64 + mt * 16 + l16) * LDST + kb;
      FragU f;
      f.u[0] = *(const uint4*)p;          // K = kb .. kb+7
      f.u[1] = *(const uint4*)(p + 16);   // K = kb+16 .. kb+23
      a[mt] = f.bf;
    }
#pragma unroll
    for (int nt = 0; nt < 2; ++nt) {
      const bf16* p = Bs + (wn * 32 + nt * 16 + l16) * LDST + kb;
      FragU f;
      f.u[0] = *(const uint4*)p;
      f.u[1] = *(const uint4*)(p + 16);
      b[nt] = f.bf;
    }
#pragma unroll
    for (int mt = 0; mt < 4; ++mt)
#pragma unroll
      for (int nt = 0; nt < 2; ++nt)
        acc[mt][nt] = __builtin_amdgcn_wmma_f32_16x16x32_bf16(
            false, a[mt], false, b[nt], (short)0, acc[mt][nt], false, false);
  }
}

// ---------------------------------------------------------------------------
// WMMA GEMM: Out[M][Nout] = A[M][K] * B[Nout][K]^T (+ A2[M][K2] * B2[Nout][K2]^T)
// 128x128 block tile, 8 waves in 2x4, each wave = 4x2 grid of 16x16 WMMA tiles.
// Double-buffered LDS fed by async-to-LDS; one barrier + one wait per K chunk.
// Tail (rank-128 correction) chunks are folded into the same pipeline.
// ---------------------------------------------------------------------------
template <bool OUT_BF16, bool DO_TAIL>
__global__ __launch_bounds__(256)
void caldera_gemm_wmma(const bf16* __restrict__ A, const bf16* __restrict__ B,
                       const bf16* __restrict__ A2, const bf16* __restrict__ B2,
                       void* __restrict__ Out, int K, int Nout, int K2) {
  __shared__ bf16 As[2][TM * LDST];
  __shared__ bf16 Bs[2][TN * LDST];

  const int m0 = blockIdx.y * TM;
  const int n0 = blockIdx.x * TN;
  const int lane = threadIdx.x & 31;
  const int wave = threadIdx.x >> 5;
  const int wm = wave >> 2;    // 0..1 : 64-row strip
  const int wn = wave & 3;     // 0..3 : 32-col strip
  const int half = lane >> 4;
  const int l16 = lane & 15;

  const int mainChunks = K / KC;
  const int nchunks = mainChunks + (DO_TAIL ? K2 / KC : 0);

  v8f acc[4][2];
#pragma unroll
  for (int mt = 0; mt < 4; ++mt)
#pragma unroll
    for (int nt = 0; nt < 2; ++nt)
      acc[mt][nt] = (v8f){0.f, 0.f, 0.f, 0.f, 0.f, 0.f, 0.f, 0.f};

  // Prologue: stage chunk 0 into buffer 0.
  issue_async_tile(A, K, m0, 0, As[0]);
  issue_async_tile(B, K, n0, 0, Bs[0]);

  for (int c = 0; c < nchunks; ++c) {
    const int buf = c & 1;
    wait_asynccnt0();     // my async-LDS writes for chunk c are complete
    __syncthreads();      // everyone's are; also fences reads of buf^1 from c-1
    if (c + 1 < nchunks) {
      const int nc = c + 1;
      if (!DO_TAIL || nc < mainChunks) {
        issue_async_tile(A, K, m0, nc * KC, As[buf ^ 1]);
        issue_async_tile(B, K, n0, nc * KC, Bs[buf ^ 1]);
      } else {
        const int k2 = (nc - mainChunks) * KC;
        issue_async_tile(A2, K2, m0, k2, As[buf ^ 1]);
        issue_async_tile(B2, K2, n0, k2, Bs[buf ^ 1]);
      }
    }
    mma_chunk(As[buf], Bs[buf], acc, wm, wn, half, l16);
  }

  // Epilogue: C/D layout => VGPR i holds row (half*8 + i), col = l16.
#pragma unroll
  for (int mt = 0; mt < 4; ++mt)
#pragma unroll
    for (int nt = 0; nt < 2; ++nt) {
      AccU u;
      u.v = acc[mt][nt];
      const int n = n0 + wn * 32 + nt * 16 + l16;
      if (OUT_BF16) {
        bf16* O = (bf16*)Out;
#pragma unroll
        for (int i = 0; i < 8; ++i) {
          int m = m0 + wm * 64 + mt * 16 + half * 8 + i;
          O[(size_t)m * Nout + n] = (bf16)u.f[i];
        }
      } else {
        float* O = (float*)Out;
#pragma unroll
        for (int i = 0; i < 8; ++i) {
          int m = m0 + wm * 64 + mt * 16 + half * 8 + i;
          O[(size_t)m * Nout + n] = u.f[i];
        }
      }
    }
}

// ---------------------------------------------------------------------------
// Host orchestration
// ---------------------------------------------------------------------------
extern "C" void kernel_launch(void* const* d_in, const int* in_sizes, int n_in,
                              void* d_out, int out_size, void* d_ws, size_t ws_size,
                              hipStream_t stream) {
  const float* x      = (const float*)d_in[0];
  const float* SU     = (const float*)d_in[1];
  const float* SV     = (const float*)d_in[2];
  const float* grid   = (const float*)d_in[3];
  const float* L_ft   = (const float*)d_in[4];
  const float* R_ft   = (const float*)d_in[5];
  const int*   Q_idxs = (const int*)d_in[6];
  const int*   R_idxs = (const int*)d_in[7];
  const int*   L_idxs = (const int*)d_in[8];
  float* out = (float*)d_out;

  // Workspace carve-up (bf16 everywhere): ~104 MiB total.
  bf16* xh   = (bf16*)d_ws;                      // [8192][4096]
  bf16* Wq   = xh + (size_t)ROWS * NCOLS;        // [4096][4096]
  bf16* Wcr  = Wq + (size_t)NCOLS * NCOLS;       // [128][4096]  rows 0-63 = W_r, 64-127 = R_ft
  bf16* WclT = Wcr + (size_t)RANKT * NCOLS;      // [4096][128]
  bf16* U    = WclT + (size_t)NCOLS * RANKT;     // [8192][128]

  // 1) xh = FWHT(x * SU) -> bf16
  caldera_fwht_pre<<<ROWS, 256, 0, stream>>>(x, SU, xh);

  // 2) dequantize / assemble operands
  caldera_dequant8<<<(NCOLS * (NCOLS / 8) + 255) / 256, 256, 0, stream>>>(
      Q_idxs, grid, Wq, 0.9f, NCOLS * (NCOLS / 8));                  // Q_SCALE
  caldera_dequant8<<<(64 * (NCOLS / 8) + 255) / 256, 256, 0, stream>>>(
      R_idxs, grid, Wcr, 1.05f, 64 * (NCOLS / 8));                   // R_SCALE
  caldera_f32_to_bf16<<<(64 * NCOLS + 255) / 256, 256, 0, stream>>>(
      R_ft, Wcr + (size_t)64 * NCOLS, 64 * NCOLS);
  caldera_build_wclt<<<(NCOLS * RANKT) / 256, 256, 0, stream>>>(
      L_idxs, grid, L_ft, WclT);

  // 3) U = xh @ Wcr^T  (8192 x 128, bf16 out)
  caldera_gemm_wmma<true, false><<<dim3(RANKT / TN, ROWS / TM), 256, 0, stream>>>(
      xh, Wcr, nullptr, nullptr, U, NCOLS, RANKT, 0);

  // 4) out = xh @ Wq^T + U @ WclT^T   (f32 out, pre-FWHT)
  caldera_gemm_wmma<false, true><<<dim3(NCOLS / TN, ROWS / TM), 256, 0, stream>>>(
      xh, Wq, U, WclT, out, NCOLS, NCOLS, RANKT);

  // 5) out = FWHT(out) * SV (in place)
  caldera_fwht_post<<<ROWS, 256, 0, stream>>>(out, SV);
}
